// HighLowPassNet_84619445666485
// MI455X (gfx1250) — compile-verified, matrix operand order checked by reading, not run on hardware
//
#include <hip/hip_runtime.h>
#include <hip/hip_bf16.h>

typedef __attribute__((ext_vector_type(16))) _Float16 v16h;
typedef __attribute__((ext_vector_type(8)))  _Float16 v8h;
typedef __attribute__((ext_vector_type(8)))  float    v8f;

#define PI2 6.28318530717958647692f

// ============================ elementwise utils ============================
__global__ void k_zero(float* p, long n) {
    long i = (long)blockIdx.x * blockDim.x + threadIdx.x;
    if (i < n) p[i] = 0.f;
}
__global__ void k_copy(float* dst, const float* src, long n) {
    long i = (long)blockIdx.x * blockDim.x + threadIdx.x;
    if (i < n) dst[i] = src[i];
}
__global__ void k_add(const float* a, const float* b, float* o, long n) {
    long i = (long)blockIdx.x * blockDim.x + threadIdx.x;
    if (i < n) o[i] = a[i] + b[i];
}
__global__ void k_r2c(const float* x, float2* c, long n) {
    long i = (long)blockIdx.x * blockDim.x + threadIdx.x;
    if (i < n) c[i] = make_float2(x[i], 0.f);
}
__global__ void k_c2r(const float2* c, float* x, long n) {
    long i = (long)blockIdx.x * blockDim.x + threadIdx.x;
    if (i < n) x[i] = c[i].x;
}
// stack [x, grad] -> NHWC C=2, written directly as f16 for the WMMA conv
__global__ void k_stack2(const float* x, const float* g, _Float16* h, long n) {
    long i = (long)blockIdx.x * blockDim.x + threadIdx.x;
    if (i < n) { h[2 * i] = (_Float16)x[i]; h[2 * i + 1] = (_Float16)g[i]; }
}

// ============================ centered 1-D FFT =============================
// fft2c = diag((-1)^k) * F * diag((-1)^n)  (N even, N/2 even). dir=+1 fwd, -1 inv (1/n scale).
__global__ void k_fft1d(float2* buf, int n, int logn, int dir, int cols) {
    __shared__ float2 sh[256];
    int line = blockIdx.x;
    int t = threadIdx.x;
    long base; int stride;
    if (cols == 0) { base = (long)line * n; stride = 1; }
    else { int b = line / n; int c = line - b * n; base = (long)b * n * n + c; stride = n; }
    float2 v = buf[base + (long)t * stride];
    float sgn = (t & 1) ? -1.f : 1.f;
    unsigned rev = __brev((unsigned)t) >> (32 - logn);
    sh[rev] = make_float2(v.x * sgn, v.y * sgn);
    __syncthreads();
    for (int st = 0; st < logn; ++st) {
        int half_ = 1 << st;
        if (t < (n >> 1)) {
            int pos = t & (half_ - 1);
            int i0 = ((t >> st) << (st + 1)) + pos;
            int i1 = i0 + half_;
            float ang = -PI2 * (float)dir * (float)pos / (float)(half_ << 1);
            float sw, cw; __sincosf(ang, &sw, &cw);
            float2 a = sh[i0], b2 = sh[i1];
            float2 bw = make_float2(b2.x * cw - b2.y * sw, b2.x * sw + b2.y * cw);
            sh[i0] = make_float2(a.x + bw.x, a.y + bw.y);
            sh[i1] = make_float2(a.x - bw.x, a.y - bw.y);
        }
        __syncthreads();
    }
    float scale = (dir < 0) ? (1.f / (float)n) : 1.f;
    float s2 = (t & 1) ? -scale : scale;
    float2 o = sh[t];
    buf[base + (long)t * stride] = make_float2(o.x * s2, o.y * s2);
}

// ======================= gather / scatter (NUFFT part) =====================
__global__ void k_scatter_y(const float* yr, const float* yi, const int* cols,
                            const int* gidx, float2* grid, int Mi, int n, int Mtot) {
    int s = blockIdx.x * blockDim.x + threadIdx.x;
    int b = blockIdx.y;
    if (s >= Mi) return;
    int col = cols[s];
    int iy = gidx[2 * s], ix = gidx[2 * s + 1];
    float* p = (float*)&grid[((long)b * n + iy) * n + ix];
    atomicAdd(p,     yr[(long)b * Mtot + col]);
    atomicAdd(p + 1, yi[(long)b * Mtot + col]);
}
__global__ void k_residual(const float2* grid, const float* yr, const float* yi,
                           const int* cols, const int* gidx, float2* res,
                           int Mi, int n, int Mtot) {
    int s = blockIdx.x * blockDim.x + threadIdx.x;
    int b = blockIdx.y;
    if (s >= Mi) return;
    int col = cols[s];
    int iy = gidx[2 * s], ix = gidx[2 * s + 1];
    float2 g = grid[((long)b * n + iy) * n + ix];
    res[(long)b * Mi + s] = make_float2(g.x - yr[(long)b * Mtot + col],
                                        g.y - yi[(long)b * Mtot + col]);
}
__global__ void k_scatter_res(const float2* res, const int* gidx, float2* grid,
                              int Mi, int n) {
    int s = blockIdx.x * blockDim.x + threadIdx.x;
    int b = blockIdx.y;
    if (s >= Mi) return;
    int iy = gidx[2 * s], ix = gidx[2 * s + 1];
    float2 v = res[(long)b * Mi + s];
    float* p = (float*)&grid[((long)b * n + iy) * n + ix];
    atomicAdd(p, v.x); atomicAdd(p + 1, v.y);
}

// ===================== frequency crop / insert (down/up) ===================
__global__ void k_crop(const float2* f, float2* low, int B_, int n) {
    int nh = n >> 1, q = n >> 2;
    long tot = (long)B_ * nh * nh;
    long i = (long)blockIdx.x * blockDim.x + threadIdx.x;
    if (i >= tot) return;
    int b = (int)(i / (nh * nh)); int rem = (int)(i - (long)b * nh * nh);
    int yy = rem / nh, xx = rem - yy * nh;
    low[i] = f[((long)b * n + yy + q) * n + xx + q];
}
__global__ void k_insert(float2* high, const float2* flow, int B_, int n) {
    int nh = n >> 1, q = n >> 2;
    long tot = (long)B_ * nh * nh;
    long i = (long)blockIdx.x * blockDim.x + threadIdx.x;
    if (i >= tot) return;
    int b = (int)(i / (nh * nh)); int rem = (int)(i - (long)b * nh * nh);
    int yy = rem / nh, xx = rem - yy * nh;
    high[((long)b * n + yy + q) * n + xx + q] = flow[i];
}

// ===================== implicit-GEMM 3x3 conv via WMMA =====================
// One wave -> 16 row-aligned pixels x 16 out-channels; K = 9*CIN padded to 32,
// fully unrolled. A layout (16-bit 16x32): lane L holds row M=L&15; element j
// holds K = (j<8 ? j : j+8) + 8*(L>=16). For CIN>=16 the +8 half-offset is a
// pure +8-channel shift folded into the base pointers, and each fragment half
// (8 consecutive K) is 8 consecutive channels of ONE 3x3 tap => one predicated
// 16B global load. Weights staged transposed [ncol][Kpad] in LDS (16B aligned)
// => each B fragment half is one ds_load_b128.
template <int CIN>
__global__ __launch_bounds__(128)
void k_conv3x3_wmma(const _Float16* __restrict__ in, const float* __restrict__ w,
                    const float* __restrict__ bias, float* __restrict__ out,
                    int ntiles, int n, int logn, int Cout, int relu) {
    constexpr int KC = 9 * CIN;
    constexpr int KPAD = (KC + 31) & ~31;
    __shared__ __align__(16) _Float16 wsh[16 * KPAD];
    int tid = threadIdx.x;
    int coutbase = blockIdx.y * 16;
    for (int idx = tid; idx < 16 * KPAD; idx += 128) {
        int ncw = idx / KPAD, k = idx - ncw * KPAD;
        wsh[idx] = (_Float16)((k < KC) ? w[(long)k * Cout + coutbase + ncw] : 0.f);
    }
    __syncthreads();

    int wave = tid >> 5, lane = tid & 31;
    int tile = blockIdx.x * 4 + wave;
    if (tile >= ntiles) return;
    int lntr = logn - 4;                         // log2(tiles per row)
    int row = tile >> lntr;                      // global row in [0, Bn*n)
    int xbase = (tile - (row << lntr)) << 4;
    int y0 = row & (n - 1);                      // row within image
    int m = lane & 15, hf = lane >> 4;
    int x0 = xbase + m;
    int hofs = hf << 3;

    long pixbase = ((long)row * n + x0) * CIN;
    const _Float16* r0 = in + pixbase - (long)n * CIN;
    const _Float16* r1 = in + pixbase;
    const _Float16* r2 = in + pixbase + (long)n * CIN;
    if (CIN >= 16) { r0 += hofs; r1 += hofs; r2 += hofs; }
    bool yok0 = y0 > 0, yok2 = y0 < n - 1;
    bool xok0 = x0 > 0, xok2 = x0 < n - 1;
    const _Float16* wl = wsh + m * KPAD + hofs;  // B base (hofs valid for all CIN)

    // 8 consecutive K starting at compile-time kbase -> one predicated 16B load
    auto loadA8 = [&](int kbase) -> v8h {
        v8h z = {};
        if (kbase >= KC) return z;
        int r = kbase / CIN, c = kbase - (kbase / CIN) * CIN;
        int dy = r / 3, dx = r - dy * 3;
        const _Float16* rp = (dy == 0) ? r0 : ((dy == 1) ? r1 : r2);
        bool ok = ((dy == 0) ? yok0 : ((dy == 2) ? yok2 : true)) &&
                  ((dx == 0) ? xok0 : ((dx == 2) ? xok2 : true));
        return ok ? *(const v8h*)(rp + (dx - 1) * CIN + c) : z;
    };

    v8f acc = {};
#pragma unroll
    for (int k0 = 0; k0 < KPAD; k0 += 32) {
        v16h av, bv;
        v8h blo = *(const v8h*)(wl + k0);
        v8h bhi = *(const v8h*)(wl + k0 + 16);
        if (CIN >= 16) {
            v8h alo = loadA8(k0);
            v8h ahi = loadA8(k0 + 16);
#pragma unroll
            for (int j = 0; j < 8; ++j) {
                av[j] = alo[j]; av[j + 8] = ahi[j];
                bv[j] = blo[j]; bv[j + 8] = bhi[j];
            }
        } else {  // CIN==2: single K-step, runs cross taps -> scalar decode
#pragma unroll
            for (int j = 0; j < 16; ++j) {
                int kk = k0 + ((j < 8) ? j : j + 8) + hofs;
                _Float16 va = (_Float16)0.f;
                if (kk < KC) {
                    int r = kk >> 1, c = kk & 1;
                    int dy = r / 3, dx = r - dy * 3;
                    const _Float16* rp = (dy == 0) ? r0 : ((dy == 1) ? r1 : r2);
                    bool ok = ((dy == 0) ? yok0 : ((dy == 2) ? yok2 : true)) &&
                              ((dx == 0) ? xok0 : ((dx == 2) ? xok2 : true));
                    if (ok) va = rp[(dx - 1) * CIN + c];
                }
                av[j] = va;
                bv[j] = (j < 8) ? blo[j] : bhi[j - 8];
            }
        }
        acc = __builtin_amdgcn_wmma_f32_16x16x32_f16(false, av, false, bv,
                                                     (short)0, acc, false, false);
    }

    float bb = bias[coutbase + m];
    long outbase = (long)row * n + xbase;
#pragma unroll
    for (int rr = 0; rr < 8; ++rr) {
        float v = acc[rr] + bb;
        if (relu) v = fmaxf(v, 0.f);
        out[(outbase + rr + hofs) * Cout + coutbase + m] = v;
    }
}

__global__ void k_conv1x1(const float* in, const float* w, const float* bias,
                          float* out, long total, int Cin) {
    long p = (long)blockIdx.x * blockDim.x + threadIdx.x;
    if (p >= total) return;
    float s = bias[0];
    for (int c = 0; c < Cin; ++c) s += in[p * Cin + c] * w[c];
    out[p] = s;
}

// ============================== batch norm =================================
__global__ void k_bn_stats(const float* x, float* stats, long total, int C) {
    __shared__ float ssum[64], ssq[64];
    int tid = threadIdx.x;
    if (tid < C) { ssum[tid] = 0.f; ssq[tid] = 0.f; }
    __syncthreads();
    long ne = total * C;
    for (long e = (long)blockIdx.x * blockDim.x + tid; e < ne;
         e += (long)gridDim.x * blockDim.x) {
        int c = (int)(e % C);
        float v = x[e];
        atomicAdd(&ssum[c], v);
        atomicAdd(&ssq[c], v * v);
    }
    __syncthreads();
    if (tid < C) { atomicAdd(&stats[tid], ssum[tid]); atomicAdd(&stats[C + tid], ssq[tid]); }
}
// normalizes in place (f32) and emits an f16 copy for the following WMMA conv.
__global__ void k_bn_apply(float* x, _Float16* xh, const float* stats, const float* g,
                           const float* be, long total, int C) {
    long e = (long)blockIdx.x * blockDim.x + threadIdx.x;
    if (e >= total * C) return;
    int c = (int)(e % C);
    float inv = 1.f / (float)total;
    float mu = stats[c] * inv;
    float var = stats[C + c] * inv - mu * mu;
    float v = g[c] * (x[e] - mu) * rsqrtf(var + 1e-3f) + be[c];
    x[e] = v;
    xh[e] = (_Float16)v;
}

// ================================ driver ===================================
extern "C" void kernel_launch(void* const* d_in, const int* in_sizes, int n_in,
                              void* d_out, int out_size, void* d_ws, size_t ws_size,
                              hipStream_t stream) {
    (void)n_in; (void)out_size; (void)ws_size;
    const int Bn = 20, Nn = 256, Mtot = 100000;
    auto Pp = [&](int blk, int it) { return (const float*)d_in[blk * 10 + it]; };
    const float* y_real = (const float*)d_in[50];
    const float* y_imag = (const float*)d_in[51];
    const int* gidx[3]; const int* colsA[3]; int Mi[3];
    for (int i = 0; i < 3; ++i) {
        gidx[i]  = (const int*)d_in[52 + i];
        colsA[i] = (const int*)d_in[55 + i];
        Mi[i]    = in_sizes[55 + i];
    }

    char* wp = (char*)d_ws;
    auto alloc = [&](size_t bytes) { void* r = wp; wp += (bytes + 255) & ~(size_t)255; return r; };
    size_t NB = (size_t)Bn * Nn * Nn;
    float2*    cgrid = (float2*)alloc(NB * 8);
    float2*    freq0 = (float2*)alloc(NB * 8);
    float2*    freq1 = (float2*)alloc((size_t)Bn * 128 * 128 * 8);
    float2*    resb  = (float2*)alloc((size_t)Bn * Mtot * 8);
    float*     xb    = (float*)alloc(NB * 4);
    float*     xinit = (float*)alloc(NB * 4);
    float*     grad  = (float*)alloc(NB * 4);
    _Float16*  hbufh = (_Float16*)alloc(NB * 2 * 2);
    float*     actA  = (float*)alloc(NB * 16 * 4);
    float*     actB  = (float*)alloc(NB * 16 * 4);
    _Float16*  actAh = (_Float16*)alloc(NB * 16 * 2);
    float*     stats = (float*)alloc(512);

    auto GB = [](long n) { return (unsigned)((n + 255) / 256); };
    auto zerof = [&](float* p, long cnt) { k_zero<<<GB(cnt), 256, 0, stream>>>(p, cnt); };
    auto fft2 = [&](float2* buf, int n, int dir) {
        int logn = __builtin_ctz((unsigned)n);
        k_fft1d<<<Bn * n, n, 0, stream>>>(buf, n, logn, dir, 0);
        k_fft1d<<<Bn * n, n, 0, stream>>>(buf, n, logn, dir, 1);
    };
    auto grad_step = [&](int i, int n) {
        long cnt = (long)Bn * n * n;
        k_r2c<<<GB(cnt), 256, 0, stream>>>(xb, cgrid, cnt);
        fft2(cgrid, n, +1);
        dim3 gs((Mi[i] + 255) / 256, Bn);
        k_residual<<<gs, 256, 0, stream>>>(cgrid, y_real, y_imag, colsA[i], gidx[i],
                                           resb, Mi[i], n, Mtot);
        zerof((float*)cgrid, cnt * 2);
        k_scatter_res<<<gs, 256, 0, stream>>>(resb, gidx[i], cgrid, Mi[i], n);
        fft2(cgrid, n, -1);
        k_c2r<<<GB(cnt), 256, 0, stream>>>(cgrid, grad, cnt);
    };
    auto conv3 = [&](const _Float16* in, const float* w, const float* bias, float* out,
                     int n, int Cin, int Cout, int relu) {
        int logn = __builtin_ctz((unsigned)n);
        int ntiles = Bn * n * (n >> 4);
        dim3 g((unsigned)((ntiles + 3) / 4), (unsigned)(Cout / 16));
        switch (Cin) {
        case 2:  k_conv3x3_wmma<2> <<<g, 128, 0, stream>>>(in, w, bias, out, ntiles, n, logn, Cout, relu); break;
        case 16: k_conv3x3_wmma<16><<<g, 128, 0, stream>>>(in, w, bias, out, ntiles, n, logn, Cout, relu); break;
        case 32: k_conv3x3_wmma<32><<<g, 128, 0, stream>>>(in, w, bias, out, ntiles, n, logn, Cout, relu); break;
        default: k_conv3x3_wmma<64><<<g, 128, 0, stream>>>(in, w, bias, out, ntiles, n, logn, Cout, relu); break;
        }
    };
    auto bnrun = [&](float* act, _Float16* acth, const float* g, const float* be,
                     int n, int C) {
        long total = (long)Bn * n * n;
        zerof(stats, 128);
        long ne = total * C;
        long blocks = (ne + 255) / 256; if (blocks > 1024) blocks = 1024;
        k_bn_stats<<<(unsigned)blocks, 256, 0, stream>>>(act, stats, total, C);
        k_bn_apply<<<GB(ne), 256, 0, stream>>>(act, acth, stats, g, be, total, C);
    };
    auto conv_block = [&](int blk, int n, int F) {
        conv3(hbufh, Pp(blk, 0), Pp(blk, 1), actA, n, 2, F, 1);
        bnrun(actA, actAh, Pp(blk, 2), Pp(blk, 3), n, F);
        conv3(actAh, Pp(blk, 4), Pp(blk, 5), actB, n, F, F, 1);
        bnrun(actB, actAh /*scratch*/, Pp(blk, 6), Pp(blk, 7), n, F);
        long total = (long)Bn * n * n;
        k_conv1x1<<<GB(total), 256, 0, stream>>>(actB, Pp(blk, 8), Pp(blk, 9), xb, total, F);
    };
    auto level_step = [&](int blk, int i, int n, int F) {
        long cnt = (long)Bn * n * n;
        grad_step(i, n);
        k_stack2<<<GB(cnt), 256, 0, stream>>>(xb, grad, hbufh, cnt);
        conv_block(blk, n, F);
    };
    auto down = [&](float2* freq, int n) {                    // xb(n) -> freq(full), xb(n/2)
        long cnt = (long)Bn * n * n;
        k_r2c<<<GB(cnt), 256, 0, stream>>>(xb, freq, cnt);
        fft2(freq, n, +1);
        long cl = cnt / 4;
        k_crop<<<GB(cl), 256, 0, stream>>>(freq, cgrid, Bn, n);
        fft2(cgrid, n / 2, -1);
        k_c2r<<<GB(cl), 256, 0, stream>>>(cgrid, xb, cl);
    };
    auto up = [&](float2* freq, int n) {                      // xb(n/2) + freq(n) -> xb(n)
        long cl = (long)Bn * (n / 2) * (n / 2);
        k_r2c<<<GB(cl), 256, 0, stream>>>(xb, cgrid, cl);
        fft2(cgrid, n / 2, +1);
        k_insert<<<GB(cl), 256, 0, stream>>>(freq, cgrid, Bn, n);
        fft2(freq, n, -1);
        long cnt = (long)Bn * n * n;
        k_c2r<<<GB(cnt), 256, 0, stream>>>(freq, xb, cnt);
    };

    // ---- x_init = real(adj_op(ys[0], idx0, 256)) ----
    zerof((float*)cgrid, (long)NB * 2);
    dim3 gs0((Mi[0] + 255) / 256, Bn);
    k_scatter_y<<<gs0, 256, 0, stream>>>(y_real, y_imag, colsA[0], gidx[0],
                                         cgrid, Mi[0], Nn, Mtot);
    fft2(cgrid, Nn, -1);
    k_c2r<<<GB((long)NB), 256, 0, stream>>>(cgrid, xinit, (long)NB);
    k_copy<<<GB((long)NB), 256, 0, stream>>>(xb, xinit, (long)NB);

    // ---- encoder ----
    level_step(0, 0, 256, 16);  down(freq0, 256);   // x: 256 -> 128
    level_step(1, 1, 128, 32);  down(freq1, 128);   // x: 128 -> 64
    // ---- bottleneck ----
    level_step(2, 2, 64, 64);
    // ---- decoder ----
    up(freq1, 128);  level_step(3, 1, 128, 32);
    up(freq0, 256);  level_step(4, 0, 256, 16);

    // ---- out = x + x_init ----
    k_add<<<GB((long)NB), 256, 0, stream>>>(xb, xinit, (float*)d_out, (long)NB);
}
